// Attention_10342281249301
// MI455X (gfx1250) — matrix operand order repaired and checked by reading, not compiled, not run on hardware
//
#include <hip/hip_runtime.h>

// Gather-attention for MI455X (gfx1250), wave32 + bf16 WMMA.
// One wave per query row n. Per head h: 1 WMMA for Q·K_nbr^T, softmax over the
// 16 neighbors with two lane-16 xor-shuffles, 2 WMMAs for softmax·V_nbr.
// Neighbor indices are wave-uniform -> lifted to SGPRs via v_readlane so all
// value gathers use uniform row-base pointers.

#define N_HEADS 8
#define HEAD_DIM 32
#define HIDDEN 256
#define KNBR 16
#define WAVES_PER_BLOCK 8

typedef __attribute__((ext_vector_type(16))) __bf16   v16bf;
typedef __attribute__((ext_vector_type(8)))  float    v8f;
typedef __attribute__((ext_vector_type(8)))  unsigned v8u;

// Pack two f32 into one dword of bf16 (lo|hi) with nearest rounding:
// +0x8000 bias then grab the high halves with a single v_perm_b32.
__device__ __forceinline__ unsigned pack2bf(float lo, float hi) {
    unsigned ulo = __builtin_bit_cast(unsigned, lo) + 0x8000u;
    unsigned uhi = __builtin_bit_cast(unsigned, hi) + 0x8000u;
    return __builtin_amdgcn_perm(uhi, ulo, 0x07060302u);
}

__global__ __launch_bounds__(WAVES_PER_BLOCK * 32)
void gather_attn_wmma(const float* __restrict__ keys,
                      const float* __restrict__ queries,
                      const float* __restrict__ values,
                      const int*   __restrict__ nidx,
                      float* __restrict__ out,
                      int n_rows)
{
    const int lane = threadIdx.x & 31;
    const int wave = threadIdx.x >> 5;
    int n = blockIdx.x * WAVES_PER_BLOCK + wave;
    if (n >= n_rows) return;                 // wave-uniform: EXEC stays all-ones
    n = __builtin_amdgcn_readfirstlane(n);   // make row index scalar

    const int half = lane >> 4;              // 0: lanes 0-15, 1: lanes 16-31
    const int col  = lane & 15;              // A row (neighbor) / out column

    // This lane's neighbor index (row `col` of the A matrix).
    const int myIdx = nidx[n * KNBR + col];

    // Wave-uniform scalar base pointers for the 16 gathered value rows.
    const float* vrow[KNBR];
    #pragma unroll
    for (int k = 0; k < KNBR; ++k) {
        unsigned s = (unsigned)__builtin_amdgcn_readlane(myIdx, k);
        vrow[k] = values + (size_t)s * HIDDEN;
    }

    const float* __restrict__ qrow = queries + (size_t)(unsigned)n * HIDDEN;
    float*       __restrict__ orow = out     + (size_t)(unsigned)n * HIDDEN;

    // 32-bit element offsets keep key addressing cheap.
    const unsigned koff = (unsigned)myIdx * HIDDEN + (unsigned)(half * 8);
    const unsigned qoff = (unsigned)(half * 8);

    // Warm L0/L2: the 8 head iterations re-read these 1KB rows in 64B slices.
    // Lanes col and col+16 together cover all 8 cache lines of each row.
    #pragma unroll
    for (int i = 0; i < 4; ++i) {
        __builtin_prefetch((const char*)(keys + (size_t)(unsigned)myIdx * HIDDEN)
                               + half * 512 + i * 128, 0, 3);
        __builtin_prefetch((const char*)(values + (size_t)(unsigned)myIdx * HIDDEN)
                               + half * 512 + i * 128, 0, 3);
    }

    const float scale = 0.17677669529663687f;   // HEAD_DIM^-0.5

    #pragma unroll 1
    for (int h = 0; h < N_HEADS; ++h) {
        const unsigned hb = (unsigned)(h * HEAD_DIM);

        // ---- A: gathered neighbor keys, 16x32 bf16 (MxK) ----
        // lane(half,col): elems 0..7 -> K = half*8+{0..7}, elems 8..15 -> +16.
        float ak[16];
        *(float4*)(ak + 0)  = *(const float4*)(keys + (size_t)(koff + hb));
        *(float4*)(ak + 4)  = *(const float4*)(keys + (size_t)(koff + hb + 4));
        *(float4*)(ak + 8)  = *(const float4*)(keys + (size_t)(koff + hb + 16));
        *(float4*)(ak + 12) = *(const float4*)(keys + (size_t)(koff + hb + 20));
        v8u au;
        #pragma unroll
        for (int r = 0; r < 8; ++r) au[r] = pack2bf(ak[2 * r], ak[2 * r + 1]);
        const v16bf A = __builtin_bit_cast(v16bf, au);

        // ---- B: q_h replicated across all 16 columns, 32x16 bf16 (KxN) ----
        // lane(half,col): elems 0..7 -> kk = half*8+{0..7}, elems 8..15 -> +16.
        float bq[16];
        *(float4*)(bq + 0)  = *(const float4*)(qrow + (size_t)(qoff + hb));
        *(float4*)(bq + 4)  = *(const float4*)(qrow + (size_t)(qoff + hb + 4));
        *(float4*)(bq + 8)  = *(const float4*)(qrow + (size_t)(qoff + hb + 16));
        *(float4*)(bq + 12) = *(const float4*)(qrow + (size_t)(qoff + hb + 20));
        v8u bu;
        #pragma unroll
        for (int r = 0; r < 8; ++r) bu[r] = pack2bf(bq[2 * r], bq[2 * r + 1]);
        const v16bf B = __builtin_bit_cast(v16bf, bu);

        // ---- scores = K_nbr · q (all 16 C columns identical) ----
        v8f c = {};
        c = __builtin_amdgcn_wmma_f32_16x16x32_bf16(
                false, A, false, B, (short)0, c, false, false);
        // lane holds scores for neighbors m = half*8 + r, r = 0..7

        // ---- softmax over the 16 neighbors ----
        float s[8];
        float mx = -3.402823466e38f;
        #pragma unroll
        for (int r = 0; r < 8; ++r) { s[r] = c[r] * scale; mx = fmaxf(mx, s[r]); }
        mx = fmaxf(mx, __shfl_xor(mx, 16, 32));
        float sum = 0.f;
        #pragma unroll
        for (int r = 0; r < 8; ++r) { s[r] = __expf(s[r] - mx); sum += s[r]; }
        sum += __shfl_xor(sum, 16, 32);
        const float inv = 1.0f / sum;

        // ---- A2: weights replicated down rows; kk>=16 zero-padded ----
        // A layout puts exactly this lane's 8 weights in elems 0..7: w[half*8+r].
        v8u wu = {0u, 0u, 0u, 0u, 0u, 0u, 0u, 0u};
        #pragma unroll
        for (int r = 0; r < 4; ++r)
            wu[r] = pack2bf(s[2 * r] * inv, s[2 * r + 1] * inv);
        const v16bf A2 = __builtin_bit_cast(v16bf, wu);

        // ---- out_h[dbase + j] = sum_k w[k] * V[idx[k], h, dbase + j] ----
        #pragma unroll
        for (int t = 0; t < 2; ++t) {
            const unsigned doff = hb + (unsigned)(t * 16 + col);
            v8u b2u = {0u, 0u, 0u, 0u, 0u, 0u, 0u, 0u};
            {
                // B column j=col, elems 0..7 -> kk = half*8+{0..7}; kk>=16 zero.
                float vv[8];
                #pragma unroll
                for (int e = 0; e < 8; ++e) vv[e] = (half ? vrow[8 + e] : vrow[e])[doff];
                #pragma unroll
                for (int r = 0; r < 4; ++r)
                    b2u[r] = pack2bf(vv[2 * r], vv[2 * r + 1]);
            }
            const v16bf B2 = __builtin_bit_cast(v16bf, b2u);
            v8f c2 = {};
            c2 = __builtin_amdgcn_wmma_f32_16x16x32_bf16(
                    false, A2, false, B2, (short)0, c2, false, false);
            // every C row equals out_h[dbase + j]; lanes 0-15 store column col
            if (half == 0) orow[doff] = c2[0];
        }
    }
}

extern "C" void kernel_launch(void* const* d_in, const int* in_sizes, int n_in,
                              void* d_out, int out_size, void* d_ws, size_t ws_size,
                              hipStream_t stream) {
    const float* keys    = (const float*)d_in[0];
    const float* queries = (const float*)d_in[1];
    const float* values  = (const float*)d_in[2];
    const int*   nidx    = (const int*)d_in[3];
    float* out = (float*)d_out;

    const int n = in_sizes[3] / KNBR;   // neighbor_idx is (N, 16)
    const int blocks = (n + WAVES_PER_BLOCK - 1) / WAVES_PER_BLOCK;
    hipLaunchKernelGGL(gather_attn_wmma, dim3(blocks), dim3(WAVES_PER_BLOCK * 32),
                       0, stream, keys, queries, values, nidx, out, n);
}
